// RecurrentCharLM_40467181863452
// MI455X (gfx1250) — matrix operand order repaired and checked by reading, not compiled, not run on hardware
//
#include <hip/hip_runtime.h>

typedef __attribute__((ext_vector_type(16))) _Float16 v16h;
typedef __attribute__((ext_vector_type(2)))  _Float16 h2;
typedef __attribute__((ext_vector_type(2)))  __fp16   g2;   // cvt_pkrtz return type
typedef __attribute__((ext_vector_type(8)))  float    v8f;

#define Hd     128
#define VOC    256
#define BSZ    256
#define SEQ    32
#define DEPTH  100     // even -> ping-pong returns to buffer 0 each time step
#define TM     16      // batch rows per block (WMMA M)
#define NTHR   256     // 8 waves

union FragU { uint4 u[2]; v16h v; };
union PkU   { g2 g; h2 h; };

// One recurrence iteration: dst = relu(src @ W), src/dst are 16x128 f16 LDS tiles.
// One workgroup barrier; all 8 LDS loads issued before the WMMA chain.
__device__ __forceinline__ void rnn_iter(const _Float16* __restrict__ src,
                                         _Float16* __restrict__ dst,
                                         const v16h* __restrict__ wfrag,
                                         int lm, int hi, int wave)
{
    // A fragments (16x32 f16): elems 0..7 -> k = 32c+8*hi+0..7,
    // elems 8..15 -> k = 32c+16+8*hi+0..7 ; two aligned 16B LDS loads each
    uint4 u[8];
    #pragma unroll
    for (int c = 0; c < 4; ++c) {
        const _Float16* base = src + lm * Hd + 32 * c + 8 * hi;
        u[2 * c + 0] = *(const uint4*)(base);
        u[2 * c + 1] = *(const uint4*)(base + 16);
    }
    __builtin_amdgcn_sched_barrier(0);   // keep all 8 ds_load_b128 in flight

    v8f acc = {};
    #pragma unroll
    for (int c = 0; c < 4; ++c) {
        FragU f;
        f.u[0] = u[2 * c + 0];
        f.u[1] = u[2 * c + 1];
        acc = __builtin_amdgcn_wmma_f32_16x16x32_f16(
                  false, f.v, false, wfrag[c],
                  (short)0, acc, false, false);
    }

    // cvt(relu(x)) == relu(cvt(x)): pack-convert pairs, then packed f16 max
    // D layout: VGPR r -> row m = r + 8*hi, col n = 16*wave + lm
    const int ncol = 16 * wave + lm;
    const h2 z2 = {};
    #pragma unroll
    for (int r = 0; r < 4; ++r) {
        PkU c;
        c.g = __builtin_amdgcn_cvt_pkrtz(acc[2 * r], acc[2 * r + 1]);
        h2 p = c.h;
        p = __builtin_elementwise_max(p, z2);            // v_pk_max_num_f16
        dst[(2 * r + 0 + 8 * hi) * Hd + ncol] = p[0];
        dst[(2 * r + 1 + 8 * hi) * Hd + ncol] = p[1];
    }
    __syncthreads();   // writes visible AND proves all reads of src are done
}

__global__ __launch_bounds__(NTHR, 1)
void rnn_charlm_kernel(const int*   __restrict__ chars,    // (B,S)
                       const float* __restrict__ hidden,   // (B,H)
                       const float* __restrict__ embed_w,  // (VOCAB,H)
                       const float* __restrict__ Ws,       // (1,H,H) row-major [k][n]
                       const float* __restrict__ ro_w,     // (VOCAB,H) row-major [v][k]
                       const float* __restrict__ ro_b,     // (VOCAB,)
                       float*       __restrict__ logits,   // (B,S,VOCAB)
                       float*       __restrict__ h_final)  // (B,H)
{
    __shared__ _Float16 hbuf[2][TM * Hd];   // ping-pong 16x128 f16 state, 2x4KB

    const int tid  = threadIdx.x;
    const int wave = tid >> 5;            // 0..7 -> output cols 16w..16w+15
    const int lane = tid & 31;
    const int hi   = (lane >> 4) & 1;     // lane half (ISA fragment layouts)
    const int lm   = lane & 15;
    const int b0   = blockIdx.x * TM;     // first batch row of this tile

    // ---- W fragments (B operand, 32x16 per K-chunk), resident all 3200 iters ----
    // B elem i: k = 32c + 8*hi + (i&7) + 16*(i>>3), n = 16*wave + lm ; W[k][n] = Ws[k*H + n]
    v16h wfrag[4];
    {
        const int n = 16 * wave + lm;
        for (int c = 0; c < 4; ++c) {
            v16h t;
            const int kb = 32 * c + 8 * hi;
            #pragma unroll
            for (int i = 0; i < 16; ++i)
                t[i] = (_Float16)Ws[(kb + (i & 7) + ((i >> 3) << 4)) * Hd + n];
            wfrag[c] = t;
        }
    }

    // ---- readout fragments: wave handles vocab tiles jt = 2*wave, 2*wave+1 ----
    // B[k][v] = ro_w[v][k]; elem i: k = 32c + 8*hi + (i&7) + 16*(i>>3) (table order), v = 16*jt + lm
    v16h rfrag[2][4];
    float bias[2];
    for (int j = 0; j < 2; ++j) {
        const int v = 16 * (2 * wave + j) + lm;
        bias[j] = ro_b[v];
        for (int c = 0; c < 4; ++c) {
            v16h t;
            const int kb = 32 * c + 8 * hi;
            #pragma unroll
            for (int i = 0; i < 16; ++i)
                t[i] = (_Float16)ro_w[v * Hd + kb + (i & 7) + ((i >> 3) << 4)];
            rfrag[j][c] = t;
        }
    }

    // ---- init h tile (buffer 0) from 'hidden' input ----
    for (int i = tid; i < TM * Hd; i += NTHR) {
        const int m = i >> 7, k = i & 127;
        hbuf[0][i] = (_Float16)hidden[(b0 + m) * Hd + k];
    }
    __syncthreads();

    for (int t = 0; t < SEQ; ++t) {
        // h += embedding of this step's char (per batch row); h lives in buffer 0
        for (int i = tid; i < TM * Hd; i += NTHR) {
            const int m = i >> 7, k = i & 127;
            const int ch = chars[(b0 + m) * SEQ + t];
            hbuf[0][i] = (_Float16)((float)hbuf[0][i] + embed_w[ch * Hd + k]);
        }
        __syncthreads();

        // ---- 100 sequential iterations: h = relu(h @ W); 1 barrier each ----
        for (int it = 0; it < DEPTH; it += 2) {
            rnn_iter(hbuf[0], hbuf[1], wfrag, lm, hi, wave);
            rnn_iter(hbuf[1], hbuf[0], wfrag, lm, hi, wave);
        }

        // ---- readout: logits[:,t,:] = h @ ro_w.T + ro_b (h in buffer 0) ----
        {
            uint4 u[8];
            #pragma unroll
            for (int c = 0; c < 4; ++c) {
                const _Float16* base = &hbuf[0][lm * Hd + 32 * c + 8 * hi];
                u[2 * c + 0] = *(const uint4*)(base);
                u[2 * c + 1] = *(const uint4*)(base + 16);
            }
            __builtin_amdgcn_sched_barrier(0);
            v16h afrag[4];
            #pragma unroll
            for (int c = 0; c < 4; ++c) {
                FragU f;
                f.u[0] = u[2 * c + 0];
                f.u[1] = u[2 * c + 1];
                afrag[c] = f.v;
            }
            #pragma unroll
            for (int j = 0; j < 2; ++j) {
                v8f acc = {};
                #pragma unroll
                for (int c = 0; c < 4; ++c)
                    acc = __builtin_amdgcn_wmma_f32_16x16x32_f16(
                              false, afrag[c], false, rfrag[j][c],
                              (short)0, acc, false, false);
                const int v = 16 * (2 * wave + j) + lm;
                #pragma unroll
                for (int r = 0; r < 8; ++r) {
                    const int m = r + 8 * hi;
                    logits[((b0 + m) * SEQ + t) * VOC + v] = acc[r] + bias[j];
                }
            }
        }
        __syncthreads();  // readout reads done before next step's embed-add writes
    }

    // ---- h_final (f32) from buffer 0 ----
    for (int i = tid; i < TM * Hd; i += NTHR) {
        const int m = i >> 7, k = i & 127;
        h_final[(b0 + m) * Hd + k] = (float)hbuf[0][i];
    }
}

extern "C" void kernel_launch(void* const* d_in, const int* in_sizes, int n_in,
                              void* d_out, int out_size, void* d_ws, size_t ws_size,
                              hipStream_t stream) {
    const int*   chars   = (const int*)  d_in[0];
    const float* hidden  = (const float*)d_in[1];
    const float* embed_w = (const float*)d_in[2];
    const float* Ws      = (const float*)d_in[3];
    const float* ro_w    = (const float*)d_in[4];
    const float* ro_b    = (const float*)d_in[5];

    float* logits  = (float*)d_out;                       // B*S*VOCAB
    float* h_final = (float*)d_out + (size_t)BSZ * SEQ * VOC;

    dim3 grid(BSZ / TM);   // 16 blocks, one 16-row batch tile each
    dim3 block(NTHR);      // 8 waves of 32
    rnn_charlm_kernel<<<grid, block, 0, stream>>>(chars, hidden, embed_w, Ws,
                                                  ro_w, ro_b, logits, h_final);
}
